// SpatialAttentionLayer_24412594111214
// MI455X (gfx1250) — matrix operand order repaired and checked by reading, compiled, not run on hardware
//
#include <hip/hip_runtime.h>
#include <stdint.h>

// ---------------------------------------------------------------------------
// SpatialAttentionLayer fused flash-attention for MI455X (gfx1250, wave32).
// Fused path => ~10MB HBM traffic; compute via v_wmma_f32_16x16x32_f16.
// K-tile staging via Tensor Data Mover (tensor_load_to_lds + s_wait_tensorcnt);
// DPP row_ror reductions; 128 blocks x 4 waves for WGP coverage.
// ---------------------------------------------------------------------------

#define N_ROWS  8192
#define IN_DIM  128
#define HID     64
#define OUT_DIM 128
#define XLD     131   // X row stride = IN_DIM + P_DIM

#define ATT_WAVES 4                    // waves per attention block
#define ATT_ROWS  (ATT_WAVES * 16)     // rows per attention block

typedef __attribute__((ext_vector_type(16))) _Float16 v16h;
typedef __attribute__((ext_vector_type(8)))  float    v8f;
typedef __attribute__((ext_vector_type(4)))  uint32_t v4u;
typedef __attribute__((ext_vector_type(8)))  int      v8i;
typedef __attribute__((ext_vector_type(4)))  int      v4i;

union FragH { v16h v; uint32_t u[8]; };
union FragF { v8f  v; float    f[8]; };

#if __has_builtin(__builtin_amdgcn_tensor_load_to_lds) && \
    __has_builtin(__builtin_amdgcn_s_wait_tensorcnt)
#define HAVE_TDM 1
#else
#define HAVE_TDM 0
#endif

#if __has_builtin(__builtin_amdgcn_update_dpp)
#define HAVE_DPP 1
#else
#define HAVE_DPP 0
#endif

static __device__ __forceinline__ uint32_t pkh(float a, float b) {
  union { _Float16 h[2]; uint32_t u; } x;
  x.h[0] = (_Float16)a; x.h[1] = (_Float16)b;
  return x.u;
}

static __device__ __forceinline__ int lane_id() {
  return (int)__builtin_amdgcn_mbcnt_lo(~0u, 0u);
}
static __device__ __forceinline__ float bperm(float v, int srcLane) {
  return __int_as_float(__builtin_amdgcn_ds_bpermute(srcLane << 2, __float_as_int(v)));
}

// --- reductions across the 16 n-lanes of a wave half ---
#if HAVE_DPP
#define ROR16(x, n) __int_as_float(__builtin_amdgcn_update_dpp( \
    0, __float_as_int(x), 0x120 | (n), 0xf, 0xf, true))
static __device__ __forceinline__ float red_max16(float x, int li) {
  (void)li;
  x = fmaxf(x, ROR16(x, 1));
  x = fmaxf(x, ROR16(x, 2));
  x = fmaxf(x, ROR16(x, 4));
  x = fmaxf(x, ROR16(x, 8));
  return x;
}
static __device__ __forceinline__ float red_sum16(float x, int li) {
  (void)li;
  x += ROR16(x, 1);
  x += ROR16(x, 2);
  x += ROR16(x, 4);
  x += ROR16(x, 8);
  return x;
}
#else
static __device__ __forceinline__ float red_max16(float x, int li) {
  for (int m = 1; m < 16; m <<= 1) x = fmaxf(x, bperm(x, li ^ m));
  return x;
}
static __device__ __forceinline__ float red_sum16(float x, int li) {
  for (int m = 1; m < 16; m <<= 1) x += bperm(x, li ^ m);
  return x;
}
#endif

#if HAVE_TDM
// TDM load of one K tile: 32 rows x 64 f16 contiguous in memory (4KB), into
// LDS with +1 dword pad every 32 dwords (row stride 66 halves) via D# pad
// fields.  6-arg builtin form on this toolchain.
static __device__ __forceinline__ void tdm_load_k_tile(const _Float16* gsrc,
                                                       uint32_t lds_byte_addr) {
  const uint64_t ga  = (uint64_t)(uintptr_t)gsrc;
  const uint32_t td0 = 0x100000u;            // generous tensor_dim0 (no OOB)
  v4u g0;
  g0[0] = 1u;                                 // count=1, user descriptor
  g0[1] = lds_byte_addr;                      // lds_addr
  g0[2] = (uint32_t)ga;                       // global_addr[31:0]
  g0[3] = (uint32_t)((ga >> 32) & 0x01FFFFFFu) | (2u << 30); // addr[56:32], type=2
  v8i g1;
  g1[0] = (int)((1u << 16)                    // data_size = 2 bytes
              | (1u << 20)                    // pad_enable
              | (4u << 22));                  // pad_interval: every 32 dwords
                                              // pad_amount encoded 0 => 1 dword
  g1[1] = (int)((td0 & 0xFFFFu) << 16);       // tensor_dim0[15:0]  @ bits 63:48
  g1[2] = (int)(((td0 >> 16) & 0xFFFFu)       // tensor_dim0[31:16] @ bits 79:64
              | (1u << 16));                  // tensor_dim1 = 1
  g1[3] = (int)(2048u << 16);                 // tile_dim0 = 2048 elements
  g1[4] = 1;                                  // tile_dim1 = 1, tile_dim2 = 0
  g1[5] = (int)td0;                           // tensor_dim0_stride[31:0]
  g1[6] = 0;
  g1[7] = 0;
  v4i z4 = {0, 0, 0, 0};
  v8i z8 = {0, 0, 0, 0, 0, 0, 0, 0};
  __builtin_amdgcn_tensor_load_to_lds(g0, g1, z4, z4, z8, 0);
}
#endif

// ---------------------------------------------------------------------------
// Kernel 0: one-shot weight swizzle.  Wt[q][c] = pack_f16(W[2q][c], W[2q+1][c])
// ---------------------------------------------------------------------------
__global__ __launch_bounds__(256) void wprep_kernel(
    const float* __restrict__ Wq, const float* __restrict__ Wk,
    const float* __restrict__ Wv,
    uint32_t* __restrict__ Wtq, uint32_t* __restrict__ Wtk,
    uint32_t* __restrict__ Wtv)
{
  int idx = blockIdx.x * 256 + threadIdx.x;   // 0..16383
  const float* W; uint32_t* Wt; int ncol, i;
  if (idx < 4096)      { W = Wq; Wt = Wtq; ncol = 64;  i = idx; }
  else if (idx < 8192) { W = Wk; Wt = Wtk; ncol = 64;  i = idx - 4096; }
  else                 { W = Wv; Wt = Wtv; ncol = 128; i = idx - 8192; }
  int q = i / ncol, c = i % ncol;
  Wt[i] = pkh(W[(size_t)(2 * q) * ncol + c], W[(size_t)(2 * q + 1) * ncol + c]);
}

// ---------------------------------------------------------------------------
// Kernel 1: QKV projection (WMMA f16).  NCOL is a template constant so the 8
// B-fragment loads become immediate-offset global loads (no address chains).
// ---------------------------------------------------------------------------
template <int NCOL>
static __device__ __forceinline__ void proj_tile(
    const FragH a[4], const uint32_t* __restrict__ Wt,
    const float* __restrict__ bias, _Float16* __restrict__ Outp,
    int cbase, int rowbase, int h, int ln)
{
  v8f acc = {};
  const uint32_t* wbase = Wt + (size_t)(8 * h) * NCOL + cbase + ln;
#pragma unroll
  for (int ks = 0; ks < 4; ++ks) {
    FragH b;
#pragma unroll
    for (int p = 0; p < 8; ++p)
      b.u[p] = wbase[(size_t)(16 * ks + p) * NCOL];   // constant imm offsets
    acc = __builtin_amdgcn_wmma_f32_16x16x32_f16(false, a[ks].v, false, b.v,
                                                 (short)0, acc, false, false);
  }
  float bb = bias[cbase + ln];
#pragma unroll
  for (int v = 0; v < 8; ++v) {
    int row = rowbase + v + 8 * h;
    Outp[(size_t)row * NCOL + cbase + ln] = (_Float16)(acc[v] + bb);
  }
}

__global__ __launch_bounds__(128) void qkv_kernel(
    const float* __restrict__ X,
    const uint32_t* __restrict__ Wtq, const float* __restrict__ bq,
    const uint32_t* __restrict__ Wtk, const float* __restrict__ bk,
    const uint32_t* __restrict__ Wtv, const float* __restrict__ bv,
    _Float16* __restrict__ Qh, _Float16* __restrict__ Kh,
    _Float16* __restrict__ Vh, float* __restrict__ Pp)
{
  __shared__ _Float16 sX[16 * 130];

  const int tid = threadIdx.x;
  const int w   = tid >> 5;
  const int li  = lane_id();
  const int h   = li >> 4;
  const int ln  = li & 15;
  const int rowbase = blockIdx.x * 16;

#pragma unroll
  for (int i = 0; i < 16; ++i) {
    int t = tid + 128 * i;
    int r = t >> 7, c = t & 127;
    sX[r * 130 + c] = (_Float16)X[(size_t)(rowbase + r) * XLD + c];
  }
  if (tid < 48) {
    int r = tid / 3, d = tid % 3;
    Pp[(size_t)(rowbase + r) * 3 + d] = X[(size_t)(rowbase + r) * XLD + IN_DIM + d];
  }
  __syncthreads();

  // A fragments (shared by all 4 output tiles of this wave)
  FragH a[4];
#pragma unroll
  for (int ks = 0; ks < 4; ++ks)
#pragma unroll
    for (int p = 0; p < 8; ++p) {
      int koff = 2 * p + 16 * h + 32 * ks;
      a[ks].u[p] = *(const uint32_t*)&sX[ln * 130 + koff];
    }

#pragma unroll
  for (int i = 0; i < 4; ++i) {
    const int tt = w * 4 + i;                 // wave-uniform tile id 0..15
    if (tt < 4)
      proj_tile<64>(a, Wtq, bq, Qh, tt * 16, rowbase, h, ln);
    else if (tt < 8)
      proj_tile<64>(a, Wtk, bk, Kh, (tt - 4) * 16, rowbase, h, ln);
    else
      proj_tile<128>(a, Wtv, bv, Vh, (tt - 8) * 16, rowbase, h, ln);
  }
}

// ---------------------------------------------------------------------------
// Kernel 2: fused spatial attention, double-buffered tiles, 1 barrier/iter.
// 128 blocks x 4 waves (64 rows/block) for WGP coverage; ~31KB LDS/block.
// ---------------------------------------------------------------------------
__global__ __launch_bounds__(128) void attn_kernel(
    const float* __restrict__ X, const float* __restrict__ log_sigma,
    const _Float16* __restrict__ Qh, const _Float16* __restrict__ Kh,
    const _Float16* __restrict__ Vh, const float* __restrict__ Pp,
    float* __restrict__ Out)
{
  __shared__ alignas(16) _Float16 sK[2][32 * 66];   // K tile, row stride 66
  __shared__ uint32_t sV[2][128 * 17];              // V tile: [col][k-pair]+pad
  __shared__ float    sP[2][96];                    // column positions
  __shared__ _Float16 sProb[ATT_WAVES][16 * 34];    // per-wave prob tile

  const int tid = threadIdx.x;
  const int w   = tid >> 5;
  const int li  = lane_id();
  const int h   = li >> 4;
  const int ln  = li & 15;
  const int rowbase = blockIdx.x * ATT_ROWS + w * 16;

  const float sig    = __expf(log_sigma[0]);
  const float sig2   = fmaxf(sig * sig, 1.0e-4f);
  const float inv2s2 = 1.0f / (2.0f * sig2);

  // Q fragments (two 32-wide K chunks); fold 4h into base => imm offsets
  FragH qa[2];
  {
    const uint32_t* qrow = (const uint32_t*)(Qh + (size_t)(rowbase + ln) * HID)
                         + 4 * h;
#pragma unroll
    for (int ks = 0; ks < 2; ++ks)
#pragma unroll
      for (int p = 0; p < 8; ++p)
        qa[ks].u[p] = qrow[((p < 4) ? p : (p + 4)) + 16 * ks];
  }

  // augmented-position A fragment: row -> [px,py,pz,|p|^2,1,0...]
  FragH pa;
  {
    const float* pr = Pp + (size_t)(rowbase + ln) * 3;
    float px = pr[0], py = pr[1], pz = pr[2];
    float pn = px * px + py * py + pz * pz;
#pragma unroll
    for (int p = 0; p < 8; ++p) pa.u[p] = 0u;
    if (h == 0) {
      pa.u[0] = pkh(px, py);
      pa.u[1] = pkh(pz, pn);
      pa.u[2] = pkh(1.0f, 0.0f);
    }
  }

  float mrow[8], srow[8];
  FragF o[8];
#pragma unroll
  for (int v = 0; v < 8; ++v) { mrow[v] = -3.0e38f; srow[v] = 0.0f; }
#pragma unroll
  for (int d = 0; d < 8; ++d)
#pragma unroll
    for (int v = 0; v < 8; ++v) o[d].f[v] = 0.0f;

  // --- tile stagers (128 threads) ---
  auto stage_v = [&](int j0, int b) {
    const uint32_t* vg = (const uint32_t*)(Vh + (size_t)j0 * OUT_DIM);
    uint16_t* v16 = (uint16_t*)sV[b];
#pragma unroll
    for (int i = 0; i < 16; ++i) {              // 2048 dwords -> [c][kpair]
      int t = tid + 128 * i;
      int r = t >> 6, cp = t & 63;
      uint32_t wv = vg[t];
      int half = r & 1, q2 = (r >> 1) * 2;
      v16[(2 * cp)     * 34 + q2 + half] = (uint16_t)(wv & 0xffffu);
      v16[(2 * cp + 1) * 34 + q2 + half] = (uint16_t)(wv >> 16);
    }
    if (tid < 96) sP[b][tid] = Pp[(size_t)j0 * 3 + tid];
  };
#if !HAVE_TDM
  auto stage_k = [&](int j0, int b) {
    const uint32_t* kg = (const uint32_t*)(Kh + (size_t)j0 * HID);
#pragma unroll
    for (int i = 0; i < 8; ++i) {
      int t = tid + 128 * i;
      int r = t >> 5, c = t & 31;
      ((uint32_t*)&sK[b][r * 66])[c] = kg[t];
    }
  };
#endif

  // --- prologue: stage tile 0 into buffer 0 ---
#if HAVE_TDM
  if (w == 0) tdm_load_k_tile(Kh, (uint32_t)(uintptr_t)&sK[0][0]);
#else
  stage_k(0, 0);
#endif
  stage_v(0, 0);

  for (int j0 = 0; j0 < N_ROWS; j0 += 32) {
    const int b = (j0 >> 5) & 1;
#if HAVE_TDM
    if (w == 0) __builtin_amdgcn_s_wait_tensorcnt(0);
#endif
    __syncthreads();                             // tiles for j0 ready; b^1 free
    if (j0 + 32 < N_ROWS) {                      // prefetch next tile (overlaps)
#if HAVE_TDM
      if (w == 0)
        tdm_load_k_tile(Kh + (size_t)(j0 + 32) * HID,
                        (uint32_t)(uintptr_t)&sK[b ^ 1][0]);
#else
      stage_k(j0 + 32, b ^ 1);
#endif
      stage_v(j0 + 32, b ^ 1);
    }

    // ---- two 16x16 score tiles: QK^T (2 WMMA) + dist^2 (1 WMMA) each ----
    FragF sc[2];
#pragma unroll
    for (int t = 0; t < 2; ++t) {
      v8f acc = {};
#pragma unroll
      for (int ks = 0; ks < 2; ++ks) {
        FragH bk;                                // B(k,n) = K[j0+16t+n][k]
        const int rk = 16 * t + ln;
#pragma unroll
        for (int p = 0; p < 8; ++p)
          bk.u[p] = *(const uint32_t*)&sK[b][rk * 66 + 8 * h + 2 * p + ((p < 4) ? 0 : 8) + 32 * ks];
        acc = __builtin_amdgcn_wmma_f32_16x16x32_f16(false, qa[ks].v, false, bk.v,
                                                     (short)0, acc, false, false);
      }
      FragH b2;                                  // B(:,n)=[-2q,1,|q|^2,0...]
      const float* pc = sP[b] + (size_t)(16 * t + ln) * 3;
      float qx = pc[0], qy = pc[1], qz = pc[2];
      float qn = qx * qx + qy * qy + qz * qz;
#pragma unroll
      for (int p = 0; p < 8; ++p) b2.u[p] = 0u;
      if (h == 0) {
        b2.u[0] = pkh(-2.0f * qx, -2.0f * qy);
        b2.u[1] = pkh(-2.0f * qz, 1.0f);
        b2.u[2] = pkh(qn, 0.0f);
      }
      v8f zc = {};
      v8f d2 = __builtin_amdgcn_wmma_f32_16x16x32_f16(false, pa.v, false, b2.v,
                                                      (short)0, zc, false, false);
#pragma unroll
      for (int v = 0; v < 8; ++v) {
        float dd = fmaxf(d2[v], 0.0f);
        sc[t].f[v] = acc[v] * 0.125f * __expf(-dd * inv2s2);
      }
    }

    // ---- online softmax_one update ----
    float fac[8];
#pragma unroll
    for (int v = 0; v < 8; ++v) {
      float m01  = red_max16(fmaxf(sc[0].f[v], sc[1].f[v]), li);
      float mnew = fmaxf(mrow[v], m01);
      float f  = __expf(mrow[v] - mnew);
      float p0 = __expf(sc[0].f[v] - mnew);
      float p1 = __expf(sc[1].f[v] - mnew);
      float ps = red_sum16(p0 + p1, li);
      srow[v] = srow[v] * f + ps;
      mrow[v] = mnew;
      fac[v]  = f;
      sProb[w][(v + 8 * h) * 34 + ln]      = (_Float16)p0;
      sProb[w][(v + 8 * h) * 34 + 16 + ln] = (_Float16)p1;
    }
#pragma unroll
    for (int d = 0; d < 8; ++d)
#pragma unroll
      for (int v = 0; v < 8; ++v) o[d].f[v] *= fac[v];

    // ---- P @ V ----
    FragH ap;
#pragma unroll
    for (int p = 0; p < 8; ++p) {
      int koff = 2 * p + 16 * h;
      ap.u[p] = *(const uint32_t*)&sProb[w][ln * 34 + koff];
    }
#pragma unroll
    for (int d = 0; d < 8; ++d) {
      FragH bvf;                                 // dword = {V[2q][c], V[2q+1][c]}
#pragma unroll
      for (int p = 0; p < 8; ++p)
        bvf.u[p] = sV[b][(d * 16 + ln) * 17 + (8 * h + p)];
      o[d].v = __builtin_amdgcn_wmma_f32_16x16x32_f16(false, ap.v, false, bvf.v,
                                                      (short)0, o[d].v, false, false);
    }
  }

  // ---- epilogue: H / (1 + sum) + Xf residual ----
#pragma unroll
  for (int v = 0; v < 8; ++v) srow[v] = 1.0f / (1.0f + srow[v]);
#pragma unroll
  for (int d = 0; d < 8; ++d)
#pragma unroll
    for (int v = 0; v < 8; ++v) {
      int row = rowbase + v + 8 * h;
      int col = d * 16 + ln;
      Out[(size_t)row * OUT_DIM + col] = o[d].f[v] * srow[v]
                                       + X[(size_t)row * XLD + col];
    }
}

// ---------------------------------------------------------------------------
extern "C" void kernel_launch(void* const* d_in, const int* in_sizes, int n_in,
                              void* d_out, int out_size, void* d_ws, size_t ws_size,
                              hipStream_t stream)
{
  (void)in_sizes; (void)n_in; (void)out_size; (void)ws_size;
  const float* X  = (const float*)d_in[0];
  const float* Wq = (const float*)d_in[1];
  const float* bq = (const float*)d_in[2];
  const float* Wk = (const float*)d_in[3];
  const float* bk = (const float*)d_in[4];
  const float* Wv = (const float*)d_in[5];
  const float* bv = (const float*)d_in[6];
  const float* ls = (const float*)d_in[7];
  float* Out = (float*)d_out;

  char* ws = (char*)d_ws;
  _Float16* Qh  = (_Float16*)(ws + 0);                 // 1 MB
  _Float16* Kh  = (_Float16*)(ws + 1 * 1048576);       // 1 MB
  _Float16* Vh  = (_Float16*)(ws + 2 * 1048576);       // 2 MB
  float*    Pp  = (float*)   (ws + 4194304);           // 96 KB
  uint32_t* Wtq = (uint32_t*)(ws + 4292608);           // 16 KB
  uint32_t* Wtk = (uint32_t*)(ws + 4308992);           // 16 KB
  uint32_t* Wtv = (uint32_t*)(ws + 4325376);           // 32 KB

  wprep_kernel<<<64, 256, 0, stream>>>(Wq, Wk, Wv, Wtq, Wtk, Wtv);
  qkv_kernel<<<512, 128, 0, stream>>>(X, Wtq, bq, Wtk, bk, Wtv, bv,
                                      Qh, Kh, Vh, Pp);
  attn_kernel<<<N_ROWS / ATT_ROWS, 128, 0, stream>>>(X, ls, Qh, Kh, Vh, Pp, Out);
}
